// Attention_43112881717925
// MI455X (gfx1250) — matrix operand order repaired and checked by reading, not compiled, not run on hardware
//
#include <hip/hip_runtime.h>
#include <stdint.h>

#define BB 2
#define SS 2048
#define DD 4096
#define HQ 32
#define HKV 8
#define HD 128
#define MROWS (BB*SS)            /* 4096 GEMM rows (b,s) */
#define NQ (HQ*HD)               /* 4096 */
#define NK (HKV*HD)              /* 1024 */
#define NTOT (NQ + 2*NK)         /* 6144 */
#define ATT_SCALE 0.08838834764831845f /* 1/sqrt(128) */

typedef __attribute__((ext_vector_type(16))) __bf16 bf16x16;
typedef __attribute__((ext_vector_type(8)))  __bf16 bf16x8;
typedef __attribute__((ext_vector_type(8)))  float  f32x8;
typedef __attribute__((ext_vector_type(4)))  float  f32x4;

/* GCC-style int4 vector: the async-to-LDS builtin's pointee type */
typedef int v4i_vs __attribute__((vector_size(16)));
typedef __attribute__((address_space(1))) v4i_vs* as1_b128_t;  /* global */
typedef __attribute__((address_space(3))) v4i_vs* as3_b128_t;  /* LDS    */

/* ---------------- CDNA5 async global->LDS copy (guarded) ---------------- */
#if defined(__has_builtin)
# if __has_builtin(__builtin_amdgcn_global_load_async_to_lds_b128)
#  define HAVE_ASYNC_LDS 1
# endif
# if __has_builtin(__builtin_amdgcn_s_wait_asynccnt)
#  define HAVE_WAIT_ASYNC 1
# endif
#endif
#ifndef HAVE_ASYNC_LDS
# define HAVE_ASYNC_LDS 0
#endif
#ifndef HAVE_WAIT_ASYNC
# define HAVE_WAIT_ASYNC 0
#endif

__device__ __forceinline__ void cp16(__bf16* lds_dst, const __bf16* gsrc) {
#if HAVE_ASYNC_LDS
  __builtin_amdgcn_global_load_async_to_lds_b128(
      (as1_b128_t)(gsrc),
      (as3_b128_t)(lds_dst),
      0, 0);
#else
  *(f32x4*)lds_dst = *(const f32x4*)gsrc;   /* sync fallback: vmem load + ds store */
#endif
}

__device__ __forceinline__ void async_wait() {
#if HAVE_ASYNC_LDS
# if HAVE_WAIT_ASYNC
  __builtin_amdgcn_s_wait_asynccnt(0);
# else
  asm volatile("s_wait_asynccnt 0" ::: "memory");
# endif
#endif
}

/* ---------------- small helpers ---------------- */
__device__ __forceinline__ bf16x16 cat16(bf16x8 lo, bf16x8 hi) {
  return __builtin_shufflevector(lo, hi, 0,1,2,3,4,5,6,7,8,9,10,11,12,13,14,15);
}
__device__ __forceinline__ f32x8 zero8() {
  f32x8 z;
#pragma unroll
  for (int i = 0; i < 8; ++i) z[i] = 0.0f;
  return z;
}
__device__ __forceinline__ f32x8 wmma_bf16(bf16x16 a, bf16x16 b, f32x8 c) {
  return __builtin_amdgcn_wmma_f32_16x16x32_bf16(false, a, false, b, (short)0, c, false, false);
}

/* ================= kernel 1: x (f32) -> xb (bf16) ================= */
__global__ void cvt_x_kernel(const float* __restrict__ x, __bf16* __restrict__ xb) {
  size_t i = ((size_t)blockIdx.x * blockDim.x + threadIdx.x) * 8;
  f32x4 a = *(const f32x4*)(x + i);
  f32x4 b = *(const f32x4*)(x + i + 4);
  bf16x8 o;
#pragma unroll
  for (int j = 0; j < 4; ++j) { o[j] = (__bf16)a[j]; o[4 + j] = (__bf16)b[j]; }
  *(bf16x8*)(xb + i) = o;
}

/* ====== kernel 2: gather wq|wk|wv -> transposed combined Wt[N=6144][K=4096] bf16 ====== */
__global__ void cvt_w_kernel(const float* __restrict__ wq, const float* __restrict__ wk,
                             const float* __restrict__ wv, __bf16* __restrict__ wbt) {
  size_t t = (size_t)blockIdx.x * blockDim.x + threadIdx.x;
  int n  = (int)(t / (DD / 8));
  int k0 = (int)(t % (DD / 8)) * 8;
  const float* src; int ld;
  if (n < NQ)            { src = wq + n;             ld = NQ; }
  else if (n < NQ + NK)  { src = wk + (n - NQ);      ld = NK; }
  else                   { src = wv + (n - NQ - NK); ld = NK; }
  bf16x8 o;
#pragma unroll
  for (int j = 0; j < 8; ++j) o[j] = (__bf16)src[(size_t)(k0 + j) * ld];
  *(bf16x8*)(wbt + (size_t)n * DD + k0) = o;
}

/* ====== kernel 3: fused QKV GEMM (bf16 WMMA) + RoPE, writes Qb/Kb and Vt (transposed) ====== */
__global__ void __launch_bounds__(128, 1)
qkv_rope_kernel(const __bf16* __restrict__ xb, const __bf16* __restrict__ wbt,
                const float* __restrict__ fc, const float* __restrict__ fs,
                __bf16* __restrict__ qb, __bf16* __restrict__ kb, __bf16* __restrict__ vt) {
  __shared__ __align__(16) __bf16 As[2][128][40];   /* 128 rows x 32 K (pad->40) */
  __shared__ __align__(16) __bf16 Bs[2][128][40];

  const int mtile = blockIdx.x % (MROWS / 128);
  const int ntile = blockIdx.x / (MROWS / 128);
  const int gm0 = mtile * 128, gn0 = ntile * 128;
  const int wave = threadIdx.x >> 5, lane = threadIdx.x & 31;
  const int lr = lane & 15, hi = lane >> 4;
  const int warpM = (wave & 1) * 64, warpN = (wave >> 1) * 64;

  auto load_tiles = [&](int st, int k0) {
#pragma unroll
    for (int c = threadIdx.x; c < 512; c += 128) {
      int row = c >> 2, part = c & 3;
      cp16(&As[st][row][part * 8], xb  + (size_t)(gm0 + row) * DD + k0 + part * 8);
      cp16(&Bs[st][row][part * 8], wbt + (size_t)(gn0 + row) * DD + k0 + part * 8);
    }
  };

  f32x8 acc[4][4];
#pragma unroll
  for (int i = 0; i < 4; ++i)
#pragma unroll
    for (int j = 0; j < 4; ++j) acc[i][j] = zero8();

  load_tiles(0, 0);
  async_wait();
  __syncthreads();

  for (int kt = 0; kt < DD / 32; ++kt) {
    const int st = kt & 1;
    if (kt + 1 < DD / 32) load_tiles(st ^ 1, (kt + 1) * 32);

    bf16x16 afr[4], bfr[4];
#pragma unroll
    for (int i = 0; i < 4; ++i) {
      const __bf16* p = &As[st][warpM + i * 16 + lr][hi * 8];
      afr[i] = cat16(*(const bf16x8*)p, *(const bf16x8*)(p + 16));
    }
#pragma unroll
    for (int j = 0; j < 4; ++j) {
      const __bf16* p = &Bs[st][warpN + j * 16 + lr][hi * 8];
      bfr[j] = cat16(*(const bf16x8*)p, *(const bf16x8*)(p + 16));
    }
#pragma unroll
    for (int i = 0; i < 4; ++i)
#pragma unroll
      for (int j = 0; j < 4; ++j) acc[i][j] = wmma_bf16(afr[i], bfr[j], acc[i][j]);

    async_wait();
    __syncthreads();
  }

  /* epilogue: RoPE (pairs are adjacent lanes in C layout) + scatter bf16 */
#pragma unroll
  for (int j = 0; j < 4; ++j) {
    const int nb = gn0 + warpN + j * 16;          /* uniform across wave */
    const int n_lane = nb + lr;
    const int d = n_lane & (HD - 1);
    const int region = (nb < NQ) ? 0 : ((nb < NQ + NK) ? 1 : 2);
    const int h = (region == 0) ? (n_lane >> 7)
                : (region == 1) ? ((n_lane - NQ) >> 7)
                                : ((n_lane - NQ - NK) >> 7);
#pragma unroll
    for (int i = 0; i < 4; ++i) {
#pragma unroll
      for (int r = 0; r < 8; ++r) {
        const int gm = gm0 + warpM + i * 16 + r + hi * 8;
        const int b_ = gm >> 11;          /* /2048 */
        const int s_ = gm & (SS - 1);
        float v = acc[i][j][r];
        float other = __shfl_xor(v, 1, 32);   /* all lanes execute */
        if (region == 2) {                    /* V: no rope, store transposed [b,h,d,s] */
          vt[(((size_t)b_ * HKV + h) * HD + d) * SS + s_] = (__bf16)v;
        } else {
          const size_t fidx = (size_t)s_ * (HD / 2) + (d >> 1);
          const float c = fc[fidx], sn = fs[fidx];
          const float out = (d & 1) ? (other * sn + v * c) : (v * c - other * sn);
          if (region == 0)
            qb[(((size_t)b_ * HQ + h) * SS + s_) * HD + d] = (__bf16)out;
          else
            kb[(((size_t)b_ * HKV + h) * SS + s_) * HD + d] = (__bf16)out;
        }
      }
    }
  }
}

/* ====== kernel 4: flash attention (bf16 WMMA, online softmax), non-causal ====== */
__global__ void __launch_bounds__(128, 1)
attn_kernel(const __bf16* __restrict__ qb, const __bf16* __restrict__ kb,
            const __bf16* __restrict__ vt, float* __restrict__ out) {
  __shared__ __align__(16) __bf16 Ks[2][32][136];  /* 32 keys x 128 d (pad->136) */
  __shared__ __align__(16) __bf16 Vs[2][128][40];  /* 128 d x 32 keys (pad->40)  */
  __shared__ __align__(16) __bf16 Ps[4][16][40];   /* per-wave P staging 16x32   */

  const int qt  = blockIdx.x % (SS / 64);
  const int h   = (blockIdx.x / (SS / 64)) % HQ;
  const int b   = blockIdx.x / ((SS / 64) * HQ);
  const int hkv = h >> 2;                            /* N_REP = 4 */
  const int wave = threadIdx.x >> 5, lane = threadIdx.x & 31;
  const int lr = lane & 15, hi = lane >> 4;
  const int qbase = qt * 64 + wave * 16;

  const __bf16* kbp = kb + ((size_t)b * HKV + hkv) * SS * HD;
  const __bf16* vtp = vt + ((size_t)b * HKV + hkv) * HD * SS;

  /* resident Q A-fragments: 16 rows x 128 d */
  const __bf16* qp = qb + (((size_t)b * HQ + h) * SS + (qbase + lr)) * HD;
  bf16x16 qf[4];
#pragma unroll
  for (int i = 0; i < 4; ++i) {
    const __bf16* p = qp + i * 32 + hi * 8;
    qf[i] = cat16(*(const bf16x8*)p, *(const bf16x8*)(p + 16));
  }

  auto load_kv = [&](int st, int k0) {
#pragma unroll
    for (int c = threadIdx.x; c < 512; c += 128) {
      {  /* K tile: 32 rows x 256B */
        int row = c >> 4, part = c & 15;
        cp16(&Ks[st][row][part * 8], kbp + (size_t)(k0 + row) * HD + part * 8);
      }
      {  /* V tile: 128 rows x 64B */
        int row = c >> 2, part = c & 3;
        cp16(&Vs[st][row][part * 8], vtp + (size_t)row * SS + k0 + part * 8);
      }
    }
  };

  f32x8 o[8];
#pragma unroll
  for (int j = 0; j < 8; ++j) o[j] = zero8();
  float m_[8], l_[8];
#pragma unroll
  for (int r = 0; r < 8; ++r) { m_[r] = -1e30f; l_[r] = 0.0f; }

  load_kv(0, 0);
  async_wait();
  __syncthreads();

  for (int kt = 0; kt < SS / 32; ++kt) {
    const int st = kt & 1;
    if (kt + 1 < SS / 32) load_kv(st ^ 1, (kt + 1) * 32);

    /* S = Q * K^T : two 16x16 key sub-tiles */
    f32x8 s0 = zero8(), s1 = zero8();
#pragma unroll
    for (int i = 0; i < 4; ++i) {
      const int d0 = i * 32 + hi * 8;
      const __bf16* p0 = &Ks[st][lr][d0];
      const __bf16* p1 = &Ks[st][16 + lr][d0];
      bf16x16 b0 = cat16(*(const bf16x8*)p0, *(const bf16x8*)(p0 + 16));
      bf16x16 b1 = cat16(*(const bf16x8*)p1, *(const bf16x8*)(p1 + 16));
      s0 = wmma_bf16(qf[i], b0, s0);
      s1 = wmma_bf16(qf[i], b1, s1);
    }

    /* online softmax: row r lives at (vgpr r, lane-half hi), cols across 16 lanes */
#pragma unroll
    for (int r = 0; r < 8; ++r) {
      float a0 = s0[r] * ATT_SCALE, a1 = s1[r] * ATT_SCALE;
      float mx = fmaxf(a0, a1);
      mx = fmaxf(mx, __shfl_xor(mx, 1, 32));
      mx = fmaxf(mx, __shfl_xor(mx, 2, 32));
      mx = fmaxf(mx, __shfl_xor(mx, 4, 32));
      mx = fmaxf(mx, __shfl_xor(mx, 8, 32));
      float mnew  = fmaxf(m_[r], mx);
      float alpha = __expf(m_[r] - mnew);
      float p0 = __expf(a0 - mnew), p1 = __expf(a1 - mnew);
      float rs = p0 + p1;
      rs += __shfl_xor(rs, 1, 32);
      rs += __shfl_xor(rs, 2, 32);
      rs += __shfl_xor(rs, 4, 32);
      rs += __shfl_xor(rs, 8, 32);
      l_[r] = l_[r] * alpha + rs;
      m_[r] = mnew;
#pragma unroll
      for (int j = 0; j < 8; ++j) o[j][r] *= alpha;
      Ps[wave][r + hi * 8][lr]      = (__bf16)p0;   /* C layout -> A layout via LDS */
      Ps[wave][r + hi * 8][16 + lr] = (__bf16)p1;
    }

    /* P A-fragment (16 q x 32 keys) */
    const __bf16* pp = &Ps[wave][lr][hi * 8];
    bf16x16 pf = cat16(*(const bf16x8*)pp, *(const bf16x8*)(pp + 16));

    /* O += P * V : Vt rows are d, contiguous over keys */
#pragma unroll
    for (int j = 0; j < 8; ++j) {
      const __bf16* vp = &Vs[st][j * 16 + lr][hi * 8];
      bf16x16 bv = cat16(*(const bf16x8*)vp, *(const bf16x8*)(vp + 16));
      o[j] = wmma_bf16(pf, bv, o[j]);
    }

    async_wait();
    __syncthreads();
  }

  /* finalize: O /= l, write fp32 out[b, q, h*128 + d] */
#pragma unroll
  for (int r = 0; r < 8; ++r) {
    const float inv = 1.0f / l_[r];
    const int q = qbase + r + hi * 8;
    float* op = out + ((size_t)b * SS + q) * (HQ * HD) + h * HD;
#pragma unroll
    for (int j = 0; j < 8; ++j) op[j * 16 + lr] = o[j][r] * inv;
  }
}

/* ================= host launch ================= */
extern "C" void kernel_launch(void* const* d_in, const int* in_sizes, int n_in,
                              void* d_out, int out_size, void* d_ws, size_t ws_size,
                              hipStream_t stream) {
  const float* x  = (const float*)d_in[0];
  const float* wq = (const float*)d_in[1];
  const float* wk = (const float*)d_in[2];
  const float* wv = (const float*)d_in[3];
  const float* fc = (const float*)d_in[6];
  const float* fs = (const float*)d_in[7];
  float* out = (float*)d_out;

  char* ws = (char*)d_ws;
  __bf16* xb  = (__bf16*)(ws);
  __bf16* wbt = xb + (size_t)MROWS * DD;
  __bf16* qb  = wbt + (size_t)NTOT * DD;
  __bf16* kb  = qb + (size_t)BB * HQ * SS * HD;
  __bf16* vt  = kb + (size_t)BB * HKV * SS * HD;

  cvt_x_kernel<<<(MROWS * (size_t)DD) / (8 * 256), 256, 0, stream>>>(x, xb);
  cvt_w_kernel<<<((size_t)NTOT * DD) / (8 * 256), 256, 0, stream>>>(wq, wk, wv, wbt);
  qkv_rope_kernel<<<(MROWS / 128) * (NTOT / 128), 128, 0, stream>>>(xb, wbt, fc, fs, qb, kb, vt);
  attn_kernel<<<BB * HQ * (SS / 64), 128, 0, stream>>>(qb, kb, vt, out);
}
